// Backbone_SPCONV_50809463111988
// MI455X (gfx1250) — matrix operand order repaired and checked by reading, not compiled, not run on hardware
//
#include <hip/hip_runtime.h>

typedef _Float16 half_t;
typedef __attribute__((ext_vector_type(16))) _Float16 v16h;
typedef __attribute__((ext_vector_type(8)))  _Float16 v8h;
typedef __attribute__((ext_vector_type(8)))  float    v8f;

// ---------------------------------------------------------------------------
// Weight packing: W (OIDHW f32) -> WMMA A-fragments (f16), K = tap*CIN + ci.
// A order packed here (contiguous k, B matches): lane = M + 16*half ; dword d, elem e ->
//   K = 2*d + e + 16*half
// Pack element index t = ((mtile*NKC + c)*32 + lane)*16 + (2*d+e)
// ---------------------------------------------------------------------------
__global__ void pack_weights(const float* __restrict__ W, half_t* __restrict__ Wp,
                             int CIN, int NKC, long total)
{
    long stride = (long)gridDim.x * blockDim.x;
    for (long t = (long)blockIdx.x * blockDim.x + threadIdx.x; t < total; t += stride) {
        int idx = (int)(t & 15);
        long r = t >> 4;
        int lane = (int)(r & 31); r >>= 5;
        int c = (int)(r % NKC);
        int mtile = (int)(r / NKC);
        int d = idx >> 1, e = idx & 1;
        int half_id = lane >> 4, m = lane & 15;
        int kq = 2 * d + e + 16 * half_id;
        int kk = c * 32 + kq;
        float v = 0.f;
        if (kk < CIN * 27) {
            int tap = kk / CIN, ci = kk - tap * CIN;
            int co = mtile * 16 + m;
            v = W[((long)co * CIN + ci) * 27 + tap];
        }
        Wp[t] = (half_t)v;
    }
}

// ---------------------------------------------------------------------------
// Implicit-GEMM conv3d 3x3x3 pad1 stride1 via v_wmma_f32_16x16x32_f16.
// Xh: f16 NDHWC [b][z][y][x][ci], Y: f32 NDHWC [p][co].
// One wave per 16-voxel N tile; NMT = COUT/16 M tiles per wave.
// ---------------------------------------------------------------------------
template<int CIN, int COUT, int S, int NMT>
__global__ __launch_bounds__(128)
void conv_wmma(const half_t* __restrict__ Xh, const half_t* __restrict__ Wp,
               float* __restrict__ Y)
{
    constexpr int NKC = (CIN * 27 + 31) / 32;
    const int lane = threadIdx.x & 31;
    const int wave = threadIdx.x >> 5;
    const long tile = (long)blockIdx.x * 4 + wave;
    const int n = lane & 15;
    const int half_id = lane >> 4;
    const long p = tile * 16 + n;
    const int x = (int)(p % S);
    const long t1 = p / S;
    const int y = (int)(t1 % S);
    const long t2 = t1 / S;
    const int z = (int)(t2 % S);
    const int b = (int)(t2 / S);

    v8f acc[NMT];
#pragma unroll
    for (int m = 0; m < NMT; ++m)
#pragma unroll
        for (int k = 0; k < 8; ++k) acc[m][k] = 0.f;

    for (int c = 0; c < NKC; ++c) {
        // B fragment: 16 consecutive K at column n = 16 consecutive channels
        // at one shifted voxel (CIN is a multiple of 16, chunk base mult of 16).
        int kkbase = c * 32 + half_id * 16;
        int tap = kkbase / CIN;
        int ci0 = kkbase - tap * CIN;
        v16h bfrag;
        bool ok = false;
        const half_t* src = nullptr;
        if (tap < 27) {
            int dz = tap / 9, dy = (tap / 3) % 3, dx = tap % 3;
            int zz = z + dz - 1, yy = y + dy - 1, xx = x + dx - 1;
            if ((unsigned)zz < (unsigned)S && (unsigned)yy < (unsigned)S &&
                (unsigned)xx < (unsigned)S) {
                ok = true;
                src = Xh + ((((long)(b * S + zz) * S + yy) * S + xx) * CIN + ci0);
            }
        }
        if (ok) {
            bfrag = *(const v16h*)src;          // 32B aligned contiguous load
        } else {
#pragma unroll
            for (int q = 0; q < 16; ++q) bfrag[q] = (_Float16)0.f;  // zero pad
        }
        const half_t* wbase = Wp + ((long)c * 32 + lane) * 16;
#pragma unroll
        for (int m = 0; m < NMT; ++m) {
            v16h afrag = *(const v16h*)(wbase + (long)m * NKC * 32 * 16);
            acc[m] = __builtin_amdgcn_wmma_f32_16x16x32_f16(
                false, afrag, false, bfrag, (short)0, acc[m], false, false);
        }
    }
    // C layout: vgpr r holds M=r (lanes 0-15) / M=r+8 (lanes 16-31) at N=n.
    // => each lane writes 8 consecutive couts: Y[p][m*16 + 8*half + 0..7]
    float* dst = Y + p * COUT + half_id * 8;
#pragma unroll
    for (int m = 0; m < NMT; ++m)
        *(v8f*)(dst + m * 16) = acc[m];
}

// ---------------------------------------------------------------------------
__global__ void init_mask_feats(const float* __restrict__ feats,
                                const int* __restrict__ occ,
                                float* __restrict__ mask, half_t* __restrict__ Xh)
{
    const long S3 = 262144;  // 64^3
    long i = (long)blockIdx.x * blockDim.x + threadIdx.x;
    if (i >= 2 * S3) return;
    long b = i / S3, v = i - b * S3;
    float m = (occ[i] == 0) ? 1.f : 0.f;
    mask[i] = m;
    const float* f = feats + (b * 16) * S3 + v;
    half_t* o = Xh + i * 16;
#pragma unroll
    for (int c = 0; c < 16; ++c) o[c] = (half_t)(f[(long)c * S3] * m);
}

__global__ void dilate_mask(const float* __restrict__ mi, float* __restrict__ mo, int S)
{
    long S3 = (long)S * S * S;
    long N = 2 * S3;
    long i = (long)blockIdx.x * blockDim.x + threadIdx.x;
    if (i >= N) return;
    long b = i / S3, v = i - b * S3;
    int x = (int)(v % S);
    int y = (int)((v / S) % S);
    int z = (int)(v / ((long)S * S));
    float a = 0.f;
    for (int dz = -1; dz <= 1; ++dz) {
        int zz = z + dz; if ((unsigned)zz >= (unsigned)S) continue;
        for (int dy = -1; dy <= 1; ++dy) {
            int yy = y + dy; if ((unsigned)yy >= (unsigned)S) continue;
            for (int dx = -1; dx <= 1; ++dx) {
                int xx = x + dx; if ((unsigned)xx >= (unsigned)S) continue;
                a += mi[b * S3 + ((long)zz * S + yy) * S + xx];
            }
        }
    }
    mo[i] = (a > 0.f) ? 1.f : 0.f;
}

__global__ void zero_f32(float* p, int n)
{
    int i = blockIdx.x * blockDim.x + threadIdx.x;
    if (i < n) p[i] = 0.f;
}

// Masked BN statistics: per-block LDS reduction, f32 atomics into sum/sumsq/n.
__global__ __launch_bounds__(256)
void bn_stats(const float* __restrict__ Y, const float* __restrict__ mask,
              float* __restrict__ sum, float* __restrict__ sumsq,
              float* __restrict__ nact, int C, long Np)
{
    const int cx = threadIdx.x;   // 0..31 (channel lane)
    const int py = threadIdx.y;   // 0..7  (position lane)
    const int nc = C >> 5;        // channels per thread (1..8)
    long p0 = (long)blockIdx.x * 4096;
    long pend = p0 + 4096; if (pend > Np) pend = Np;
    float s[8], q[8];
#pragma unroll
    for (int j = 0; j < 8; ++j) { s[j] = 0.f; q[j] = 0.f; }
    float mn = 0.f;
    for (long p = p0 + py; p < pend; p += 8) {
        float m = mask[p];
        if (cx == 0) mn += m;
        if (m != 0.f) {
            const float* yp = Y + p * C;
            for (int j = 0; j < nc; ++j) {
                float v = yp[cx + (j << 5)];
                s[j] += v; q[j] += v * v;
            }
        }
    }
    __shared__ float red[8][32];
    for (int j = 0; j < nc; ++j) {
        red[py][cx] = s[j]; __syncthreads();
        if (py == 0) {
            float t = 0.f;
            for (int k = 0; k < 8; ++k) t += red[k][cx];
            atomicAdd(&sum[cx + (j << 5)], t);
        }
        __syncthreads();
        red[py][cx] = q[j]; __syncthreads();
        if (py == 0) {
            float t = 0.f;
            for (int k = 0; k < 8; ++k) t += red[k][cx];
            atomicAdd(&sumsq[cx + (j << 5)], t);
        }
        __syncthreads();
    }
    red[py][cx] = (cx == 0) ? mn : 0.f; __syncthreads();
    if (py == 0 && cx == 0) {
        float t = 0.f;
        for (int k = 0; k < 8; ++k) t += red[k][0];
        atomicAdd(nact, t);
    }
}

__global__ void bn_finalize(const float* __restrict__ sum, const float* __restrict__ sumsq,
                            const float* __restrict__ nact,
                            const float* __restrict__ g, const float* __restrict__ bb,
                            float* __restrict__ scale, float* __restrict__ shift, int C)
{
    int c = blockIdx.x * blockDim.x + threadIdx.x;
    if (c >= C) return;
    float n = fmaxf(nact[0], 1.f);
    float mean = sum[c] / n;
    float var = fmaxf(sumsq[c] / n - mean * mean, 0.f);
    float sc = g[c] * rsqrtf(var + 1e-5f);
    scale[c] = sc;
    shift[c] = bb[c] - mean * sc;
}

__global__ void bn_apply(const float* __restrict__ Y, const float* __restrict__ mask,
                         const float* __restrict__ scale, const float* __restrict__ shift,
                         half_t* __restrict__ Xout, int C, long Np)
{
    long t = (long)blockIdx.x * blockDim.x + threadIdx.x;
    if (t >= Np * C) return;
    int c = (int)(t % C);
    long p = t / C;
    float v = fmaxf(Y[t] * scale[c] + shift[c], 0.f) * mask[p];
    Xout[t] = (half_t)v;
}

// Sparse avg-pool 3x3x3 stride2 pad1: writes f32 NCDHW module output and the
// f16 NDHWC input for the next module, plus the pooled mask.
__global__ void pool_kernel(const half_t* __restrict__ Xh, const float* __restrict__ mask,
                            int S, int C, half_t* __restrict__ Xout,
                            float* __restrict__ maskOut, float* __restrict__ dout)
{
    int So = S / 2;
    long So3 = (long)So * So * So;
    long N = 2 * So3;
    long i = (long)blockIdx.x * blockDim.x + threadIdx.x;
    if (i >= N) return;
    long b = i / So3, v = i - b * So3;
    int xo = (int)(v % So);
    int yo = (int)((v / So) % So);
    int zo = (int)(v / ((long)So * So));
    long S3 = (long)S * S * S;
    long nb[27]; int nn = 0; float cnt = 0.f;
    for (int kz = 0; kz < 3; ++kz) {
        int z = 2 * zo - 1 + kz; if ((unsigned)z >= (unsigned)S) continue;
        for (int ky = 0; ky < 3; ++ky) {
            int y = 2 * yo - 1 + ky; if ((unsigned)y >= (unsigned)S) continue;
            for (int kx = 0; kx < 3; ++kx) {
                int x = 2 * xo - 1 + kx; if ((unsigned)x >= (unsigned)S) continue;
                long p = b * S3 + ((long)z * S + y) * S + x;
                cnt += mask[p];
                nb[nn++] = p;
            }
        }
    }
    maskOut[i] = (cnt > 0.f) ? 1.f : 0.f;
    float inv = 1.f / fmaxf(cnt, 1.f);
    for (int c0 = 0; c0 < C; c0 += 8) {
        float a[8] = {0.f, 0.f, 0.f, 0.f, 0.f, 0.f, 0.f, 0.f};
        for (int t = 0; t < nn; ++t) {
            v8h h = *(const v8h*)(Xh + nb[t] * C + c0);
#pragma unroll
            for (int k = 0; k < 8; ++k) a[k] += (float)h[k];
        }
#pragma unroll
        for (int k = 0; k < 8; ++k) {
            float val = a[k] * inv;   // zero where cnt==0 (sum is zero)
            Xout[i * C + c0 + k] = (half_t)val;
            dout[(b * C + c0 + k) * So3 + v] = val;
        }
    }
}

// ---------------------------------------------------------------------------
// Host-side: one conv layer = conv + (optional mask dilate) + masked BN + ReLU.
// ---------------------------------------------------------------------------
template<int CIN, int COUT, int S>
static void layer_forward(const half_t* Xin, const half_t* WpL,
                          const float* g, const float* bb, float* Y,
                          const float* maskCur, float* maskNew, bool do_dilate,
                          half_t* Xout, float* stats, hipStream_t stream)
{
    const long Np = 2L * S * S * S;
    conv_wmma<CIN, COUT, S, COUT / 16>
        <<<(unsigned)(Np / 64), 128, 0, stream>>>(Xin, WpL, Y);
    const float* mUse = maskCur;
    if (do_dilate) {
        dilate_mask<<<(unsigned)((Np + 255) / 256), 256, 0, stream>>>(maskCur, maskNew, S);
        mUse = maskNew;
    }
    zero_f32<<<(2 * COUT + 1 + 255) / 256, 256, 0, stream>>>(stats, 2 * COUT + 1);
    bn_stats<<<(unsigned)((Np + 4095) / 4096), dim3(32, 8), 0, stream>>>(
        Y, mUse, stats, stats + COUT, stats + 2 * COUT, COUT, Np);
    bn_finalize<<<(COUT + 63) / 64, 64, 0, stream>>>(
        stats, stats + COUT, stats + 2 * COUT, g, bb,
        stats + 2 * COUT + 1, stats + 3 * COUT + 1, COUT);
    bn_apply<<<(unsigned)((Np * COUT + 255) / 256), 256, 0, stream>>>(
        Y, mUse, stats + 2 * COUT + 1, stats + 3 * COUT + 1, Xout, COUT, Np);
}

extern "C" void kernel_launch(void* const* d_in, const int* in_sizes, int n_in,
                              void* d_out, int out_size, void* d_ws, size_t ws_size,
                              hipStream_t stream)
{
    (void)in_sizes; (void)n_in; (void)out_size; (void)ws_size;
    const float* feats = (const float*)d_in[0];
    const int*   occ   = (const int*)d_in[1];
    const float* W[8]; const float* G[8]; const float* Bb[8];
    for (int i = 0; i < 8; ++i) {
        W[i]  = (const float*)d_in[2 + 3 * i + 0];
        G[i]  = (const float*)d_in[2 + 3 * i + 1];
        Bb[i] = (const float*)d_in[2 + 3 * i + 2];
    }

    // Workspace carve (256B aligned): packed weights, Xh ping/pong (f16 NDHWC),
    // conv output Y (f32 NDHWC), mask ping/pong, BN stats.
    auto alignup = [](size_t x) { return (x + 255) & ~(size_t)255; };
    char* w = (char*)d_ws;
    const long wpTotalHalfs = 3525632;
    half_t* Wp = (half_t*)w;      w += alignup((size_t)wpTotalHalfs * 2);
    const size_t xhHalfs = 16777216;   // 2*64^3*32 (layer-1 input, the max)
    half_t* XhA = (half_t*)w;     w += alignup(xhHalfs * 2);
    half_t* XhB = (half_t*)w;     w += alignup(xhHalfs * 2);
    float* Y = (float*)w;         w += alignup(xhHalfs * 4);
    float* maskA = (float*)w;     w += alignup((size_t)524288 * 4);
    float* maskB = (float*)w;     w += alignup((size_t)524288 * 4);
    float* stats = (float*)w;     // 4*256+1 floats max

    // Pack all weights into WMMA A-fragment layout (every call: deterministic).
    const int  cins[8]  = {16, 32, 32, 64, 64, 128, 128, 256};
    const int  couts[8] = {32, 32, 64, 64, 128, 128, 256, 256};
    const long wpOff[8] = {0, 14336, 41984, 97280, 207872, 429056, 871424, 1756160};
    for (int i = 0; i < 8; ++i) {
        int nkc = (cins[i] * 27 + 31) / 32;
        long tot = (long)couts[i] * nkc * 32;   // (COUT/16)*NKC*32lanes*16halfs
        unsigned blocks = (unsigned)((tot + 255) / 256);
        if (blocks > 4096) blocks = 4096;
        pack_weights<<<blocks, 256, 0, stream>>>(W[i], Wp + wpOff[i], cins[i], nkc, tot);
    }

    init_mask_feats<<<(2 * 262144 + 255) / 256, 256, 0, stream>>>(feats, occ, maskA, XhA);

    half_t* xin = XhA; half_t* xout = XhB;
    float* mP = maskA; float* mQ = maskB;
    float* dout = (float*)d_out;
    const long off1 = 2097152, off2 = 2621440, off3 = 2752512;

    // -------- Module 1 (S=64, 16->32->32) --------
    layer_forward<16, 32, 64>(xin, Wp + wpOff[0], G[0], Bb[0], Y, mP, mQ, true, xout, stats, stream);
    { float* t = mP; mP = mQ; mQ = t; } { half_t* t = xin; xin = xout; xout = t; }
    layer_forward<32, 32, 64>(xin, Wp + wpOff[1], G[1], Bb[1], Y, mP, mQ, false, xout, stats, stream);
    { half_t* t = xin; xin = xout; xout = t; }
    pool_kernel<<<(2 * 32768 + 127) / 128, 128, 0, stream>>>(xin, mP, 64, 32, xout, mQ, dout);
    { float* t = mP; mP = mQ; mQ = t; } { half_t* t = xin; xin = xout; xout = t; }

    // -------- Module 2 (S=32, 32->64->64) --------
    layer_forward<32, 64, 32>(xin, Wp + wpOff[2], G[2], Bb[2], Y, mP, mQ, true, xout, stats, stream);
    { float* t = mP; mP = mQ; mQ = t; } { half_t* t = xin; xin = xout; xout = t; }
    layer_forward<64, 64, 32>(xin, Wp + wpOff[3], G[3], Bb[3], Y, mP, mQ, false, xout, stats, stream);
    { half_t* t = xin; xin = xout; xout = t; }
    pool_kernel<<<(2 * 4096 + 127) / 128, 128, 0, stream>>>(xin, mP, 32, 64, xout, mQ, dout + off1);
    { float* t = mP; mP = mQ; mQ = t; } { half_t* t = xin; xin = xout; xout = t; }

    // -------- Module 3 (S=16, 64->128->128) --------
    layer_forward<64, 128, 16>(xin, Wp + wpOff[4], G[4], Bb[4], Y, mP, mQ, true, xout, stats, stream);
    { float* t = mP; mP = mQ; mQ = t; } { half_t* t = xin; xin = xout; xout = t; }
    layer_forward<128, 128, 16>(xin, Wp + wpOff[5], G[5], Bb[5], Y, mP, mQ, false, xout, stats, stream);
    { half_t* t = xin; xin = xout; xout = t; }
    pool_kernel<<<(2 * 512 + 127) / 128, 128, 0, stream>>>(xin, mP, 16, 128, xout, mQ, dout + off2);
    { float* t = mP; mP = mQ; mQ = t; } { half_t* t = xin; xin = xout; xout = t; }

    // -------- Module 4 (S=8, 128->256->256) --------
    layer_forward<128, 256, 8>(xin, Wp + wpOff[6], G[6], Bb[6], Y, mP, mQ, true, xout, stats, stream);
    { float* t = mP; mP = mQ; mQ = t; } { half_t* t = xin; xin = xout; xout = t; }
    layer_forward<256, 256, 8>(xin, Wp + wpOff[7], G[7], Bb[7], Y, mP, mQ, false, xout, stats, stream);
    { half_t* t = xin; xin = xout; xout = t; }
    pool_kernel<<<(2 * 64 + 127) / 128, 128, 0, stream>>>(xin, mP, 8, 256, xout, mQ, dout + off3);
}
